// SodiumUpsampler_37357625541240
// MI455X (gfx1250) — matrix operand order repaired and checked by reading, not compiled, not run on hardware
//
#include <hip/hip_runtime.h>
#include <hip/hip_bf16.h>

typedef __attribute__((ext_vector_type(8)))  float  v8f;
typedef __attribute__((ext_vector_type(2)))  float  v2f;
typedef __attribute__((ext_vector_type(16))) __bf16 v16bf;
typedef __attribute__((ext_vector_type(8)))  __bf16 v8bf;

#define S_IN 384
#define BATCH 64
#define EMB 256
#define HID 128
#define KPAD 260     // 257 padded to multiple of 4
#define G4H 512      // 4*HID
#define OUTE 288     // EMB + POS_DIM
#define POSD 32

__device__ __forceinline__ int iclamp(int v, int lo, int hi) {
    return v < lo ? lo : (v > hi ? hi : v);
}

// hardware-TRANS fast activations (v_exp_f32 based; exact saturation at +-inf)
__device__ __forceinline__ float fsig(float x) {
    return 1.0f / (1.0f + __expf(-x));
}
__device__ __forceinline__ float ftanh(float x) {
    float e = __expf(2.0f * x);
    return 1.0f - 2.0f / (e + 1.0f);
}

// ---------------------------------------------------------------------------
// Prep kernels
// ---------------------------------------------------------------------------
// Builds K-major (transposed) inputs XT[k][m] so GEMM A-fragment loads are
// coalesced across lanes (consecutive m).
__global__ void su_build_X(const float* __restrict__ dur, const float* __restrict__ tempo,
                           const int* __restrict__ td, const float* __restrict__ enc,
                           float* __restrict__ Xd, float* __restrict__ Xr, int M) {
    int i = blockIdx.x * blockDim.x + threadIdx.x;
    if (i >= M * KPAD) return;
    int k = i / M, m = i - k * M;       // i = k*M + m  (coalesced writes)
    int b = m & (BATCH - 1);
    float vd, vr;
    if (k == 0) {
        vd = dur[m] / tempo[b] * 60.0f;
        vr = (float)td[m];
    } else if (k <= EMB) {
        float e = enc[(size_t)m * EMB + (k - 1)];
        vd = e; vr = e;
    } else { vd = 0.0f; vr = 0.0f; }
    Xd[i] = vd; Xr[i] = vr;
}

// Wih (512,257) -> WT (260,512), zero-padded K
__global__ void su_build_WT(const float* __restrict__ W, float* __restrict__ WT) {
    int i = blockIdx.x * blockDim.x + threadIdx.x;
    if (i >= KPAD * G4H) return;
    int k = i / G4H, n = i - k * G4H;
    WT[i] = (k < 257) ? W[n * 257 + k] : 0.0f;
}

// Whh (512,128) -> bf16 B-fragment-swizzled WhhP:
//   K = kt*32 + half*16 + e  (kt = k>>5, half = (k>>4)&1, e = k&15)
//   P[((kt*2 + half) * 512 + n) * 16 + e] = bf16(Whh[n,k])
// so a lane's 16 K-values for one WMMA are a contiguous 32B LDS read.
__global__ void su_build_WhhP(const float* __restrict__ W, __bf16* __restrict__ P) {
    int i = blockIdx.x * blockDim.x + threadIdx.x;
    if (i >= HID * G4H) return;
    int k = i / G4H, n = i - k * G4H;
    int blk = (k >> 5) * 2 + ((k >> 4) & 1);
    P[((size_t)blk * G4H + n) * 16 + (k & 15)] = (__bf16)W[n * HID + k];
}

__global__ void su_build_bias(const float* __restrict__ b1, const float* __restrict__ b2,
                              float* __restrict__ bo) {
    int i = blockIdx.x * blockDim.x + threadIdx.x;
    if (i < G4H) bo[i] = b1[i] + b2[i];
}

// ---------------------------------------------------------------------------
// Input GEMM: Y[m,n] = XT[:,m] . WT[:,n] (+bias), M=24576, N=512, K=260
// f32 WMMA 16x16x4, 4 N-tiles per wave (A fragment reused 4x).
// grid=(192, 8, 4), block=256 (8 waves, wave per m-tile)
// ---------------------------------------------------------------------------
struct GemmSet {
    const float* X[4];
    const float* W[4];
    float* Y[4];
    const float* bias[4];
};

__global__ __launch_bounds__(256) void su_input_gemm(GemmSet gs, int M) {
    int z = blockIdx.z;
    const float* __restrict__ X = gs.X[z];    // XT: (KPAD, M)
    const float* __restrict__ W = gs.W[z];
    float* __restrict__ Y = gs.Y[z];
    const float* bias = gs.bias[z];
    int lane = threadIdx.x & 31, wid = threadIdx.x >> 5;
    int mt = blockIdx.x * 8 + wid;       // 0..1535
    int row0 = mt * 16;
    int col0 = blockIdx.y * 64;          // 4 tiles of 16
    int lm = lane & 15;
    int kh = (lane >> 4) * 2;
    int mb = (lane >> 4) * 8;
    v8f acc[4] = {};
    for (int k0 = 0; k0 < KPAD; k0 += 4) {
        v2f a;
        const float* xr = X + (size_t)(k0 + kh) * M + row0 + lm;
        a.x = xr[0];                     // coalesced across lanes (consecutive m)
        a.y = xr[M];
        const float* wr = W + (size_t)(k0 + kh) * G4H + col0 + lm;
#pragma unroll
        for (int j = 0; j < 4; ++j) {
            v2f bb;
            bb.x = wr[j * 16];
            bb.y = wr[j * 16 + G4H];
            acc[j] = __builtin_amdgcn_wmma_f32_16x16x4_f32(false, a, false, bb, (short)0,
                                                           acc[j], false, false);
        }
    }
#pragma unroll
    for (int j = 0; j < 4; ++j) {
        float bv = bias ? bias[col0 + j * 16 + lm] : 0.0f;
        for (int v = 0; v < 8; ++v)
            Y[(size_t)(row0 + mb + v) * G4H + col0 + j * 16 + lm] = acc[j][v] + bv;
    }
}

// ---------------------------------------------------------------------------
// LSTM recurrence: one block (1024 thr = 32 waves) per direction.
// All per-step state in LDS: WhhP bf16 (128KB) + G f32 (128KB) + h bf16 (16KB).
// Per step: G = xg[t] + h @ Whh^T via bf16 WMMA (16x16x32), then gates.
// Next step's xg lines are prefetched into WGP$ during the gate phase.
// ---------------------------------------------------------------------------
struct RecSet {
    const float* xg[4];
    const __bf16* whhP[4];
    float* outb[4];
    int rev[4];
};

#define REC_SMEM (128 * 1024 + 128 * 1024 + 16 * 1024)

__global__ __launch_bounds__(1024) void su_lstm_rec(RecSet rs, const int* __restrict__ lens) {
    int dir = blockIdx.x;
    const float* __restrict__ xg = rs.xg[dir];
    const __bf16* __restrict__ wsrc = rs.whhP[dir];
    float* __restrict__ outb = rs.outb[dir];
    int rev = rs.rev[dir];

    extern __shared__ char smem[];
    float*  Gs   = (float*)smem;                       // 64 x 512 f32 (128KB)
    __bf16* whhP = (__bf16*)(smem + 128 * 1024);       // swizzled Whh bf16 (128KB)
    __bf16* hsb  = (__bf16*)(smem + 256 * 1024);       // 64 x 128 h bf16 (16KB)

    int tid = threadIdx.x;
    // stage Whh into LDS (65536 bf16), init h
    for (int i = tid; i < HID * G4H; i += 1024) whhP[i] = wsrc[i];
    for (int i = tid; i < BATCH * HID; i += 1024) hsb[i] = (__bf16)0.0f;
    float creg[8];
    for (int r = 0; r < 8; ++r) creg[r] = 0.0f;
    // warm WGP$ with step-0 xg lines
    {
        int pb_ = tid >> 4;
        int pn = (tid & 15) * 32;
        int ss0 = rev ? iclamp(lens[pb_] - 1, 0, S_IN - 1) : 0;
        __builtin_prefetch(xg + (size_t)(ss0 * BATCH + pb_) * G4H + pn, 0, 3);
    }
    __syncthreads();

    int lane = tid & 31, wid = tid >> 5;
    int lm = lane & 15;
    int half = lane >> 4;          // K half select
    int mb = half * 8;             // C row base

    for (int t = 0; t < S_IN; ++t) {
        // ---- GEMM phase: 4x32 tiles of (64x512), 4 tiles per wave ----
        for (int q = 0; q < 4; ++q) {
            int tile = wid + 32 * q;
            int mt = tile & 3, nt = tile >> 2;
            int row0 = mt * 16, col0 = nt * 16;
            v8f acc;
#pragma unroll
            for (int v = 0; v < 8; ++v) {
                int b = row0 + mb + v;
                int ss = rev ? iclamp(lens[b] - 1 - t, 0, S_IN - 1) : t;
                acc[v] = xg[(size_t)(ss * BATCH + b) * G4H + col0 + lm];
            }
#pragma unroll
            for (int kt = 0; kt < 4; ++kt) {       // K = 128 in chunks of 32
                int k0 = kt * 32;
                // A fragment: h rows, one contiguous 32B chunk per lane
                const v8bf* pa = (const v8bf*)(hsb + (row0 + lm) * HID + k0 + half * 16);
                v8bf a0 = pa[0];
                v8bf a1 = pa[1];                   // +8 elements
                v16bf af = __builtin_shufflevector(a0, a1, 0, 1, 2, 3, 4, 5, 6, 7,
                                                   8, 9, 10, 11, 12, 13, 14, 15);
                // B fragment: one contiguous 32B chunk per lane (pre-swizzled)
                const v8bf* pb = (const v8bf*)(whhP +
                                  (((size_t)kt * 2 + half) * G4H + col0 + lm) * 16);
                v8bf b0 = pb[0];
                v8bf b1 = pb[1];
                v16bf bf = __builtin_shufflevector(b0, b1, 0, 1, 2, 3, 4, 5, 6, 7,
                                                   8, 9, 10, 11, 12, 13, 14, 15);
                acc = __builtin_amdgcn_wmma_f32_16x16x32_bf16(false, af, false, bf,
                                                              (short)0, acc, false, false);
            }
#pragma unroll
            for (int v = 0; v < 8; ++v)
                Gs[(row0 + mb + v) * G4H + col0 + lm] = acc[v];
        }
        __syncthreads();
        // ---- gate phase: 8192 cells, 8 per thread ----
        for (int r = 0; r < 8; ++r) {
            int idx = tid + 1024 * r;
            int b = idx >> 7, j = idx & (HID - 1);
            float gi = Gs[b * G4H + j];
            float gf = Gs[b * G4H + HID + j];
            float gg = Gs[b * G4H + 2 * HID + j];
            float go = Gs[b * G4H + 3 * HID + j];
            float cn = fsig(gf) * creg[r] + fsig(gi) * ftanh(gg);
            float hn = fsig(go) * ftanh(cn);
            bool msk = t < lens[b];
            float hprev = (float)hsb[idx];
            creg[r] = msk ? cn : creg[r];
            hsb[idx] = (__bf16)(msk ? hn : hprev);
            outb[(size_t)(t * BATCH + b) * HID + j] = msk ? hn : 0.0f;
        }
        // ---- prefetch next step's xg (1024 x 128B lines) into WGP$ ----
        if (t + 1 < S_IN) {
            int pb_ = tid >> 4;
            int pn = (tid & 15) * 32;
            int ssn = rev ? iclamp(lens[pb_] - 2 - t, 0, S_IN - 1) : (t + 1);
            __builtin_prefetch(xg + (size_t)(ssn * BATCH + pb_) * G4H + pn, 0, 3);
        }
        __syncthreads();
    }
}

// ---------------------------------------------------------------------------
// Projections: pred_durations (relu) and pred_ranges (softplus)
// ---------------------------------------------------------------------------
__global__ void su_proj(const float* __restrict__ of_d, const float* __restrict__ or_d,
                        const float* __restrict__ of_r, const float* __restrict__ or_r,
                        const float* __restrict__ dW, const float* __restrict__ rW,
                        const float* __restrict__ rb, const int* __restrict__ lens,
                        float* __restrict__ pd, float* __restrict__ ranges) {
    int i = blockIdx.x * blockDim.x + threadIdx.x;
    if (i >= S_IN * BATCH) return;
    int s = i >> 6, b = i & 63;
    int sr = iclamp(lens[b] - 1 - s, 0, S_IN - 1);
    size_t fo = (size_t)(s * BATCH + b) * HID;
    size_t ro = (size_t)(sr * BATCH + b) * HID;
    float ad = 0.0f, ar = 0.0f;
    for (int j = 0; j < HID; ++j) {
        ad += of_d[fo + j] * dW[j] + or_d[ro + j] * dW[HID + j];
        ar += of_r[fo + j] * rW[j] + or_r[ro + j] * rW[HID + j];
    }
    pd[i] = fmaxf(ad, 0.0f);
    float x = ar + rb[0];
    ranges[i] = (x > 20.0f) ? x : log1pf(expf(x));
}

// ---------------------------------------------------------------------------
// Per-batch cumsum scans (centers, int csum, totals, seq_lengths)
// ---------------------------------------------------------------------------
__global__ void su_scan(const int* __restrict__ td, float* __restrict__ centersT,
                        int* __restrict__ csumT, int* __restrict__ total,
                        float* __restrict__ seq_len) {
    int b = blockIdx.x * blockDim.x + threadIdx.x;
    if (b >= BATCH) return;
    float cf = 0.0f; int ci = 0;
    for (int s = 0; s < S_IN; ++s) {
        int dv = td[s * BATCH + b];
        float d = (float)dv;
        cf += d;
        centersT[b * S_IN + s] = cf - 0.5f * d;
        ci += dv;
        csumT[b * S_IN + s] = ci;
    }
    total[b] = ci;
    seq_len[b] = cf;
}

__global__ void su_rng_prep(const float* __restrict__ ranges, float* __restrict__ invT,
                            float* __restrict__ lgT) {
    int i = blockIdx.x * blockDim.x + threadIdx.x;
    if (i >= S_IN * BATCH) return;
    int s = i >> 6, b = i & 63;
    float r = ranges[i];
    invT[b * S_IN + s] = 1.0f / r;
    lgT[b * S_IN + s] = logf(r);
}

// ---------------------------------------------------------------------------
// Gaussian softmax over s, written directly into the weights output (S,B,T)
// ---------------------------------------------------------------------------
__global__ void su_softmax(const float* __restrict__ centersT, const float* __restrict__ invT,
                           const float* __restrict__ lgT, float* __restrict__ wout, int T) {
    int b = blockIdx.y;
    int t = blockIdx.x * blockDim.x + threadIdx.x;
    if (t >= T) return;
    const float C0 = 0.91893853320467274f;  // 0.5*log(2*pi)
    const float* cT = centersT + b * S_IN;
    const float* iT = invT + b * S_IN;
    const float* gT = lgT + b * S_IN;
    float ft = (float)t;
    float mx = -3.4e38f, sum = 0.0f;
    for (int s = 0; s < S_IN; ++s) {
        float z = (ft - cT[s]) * iT[s];
        float lp = -0.5f * z * z - gT[s] - C0;
        if (lp > mx) { sum = sum * __expf(mx - lp) + 1.0f; mx = lp; }
        else sum += __expf(lp - mx);
    }
    float inv = 1.0f / sum;
    for (int s = 0; s < S_IN; ++s) {
        float z = (ft - cT[s]) * iT[s];
        float lp = -0.5f * z * z - gT[s] - C0;
        wout[(size_t)(s * BATCH + b) * T + t] = __expf(lp - mx) * inv;
    }
}

// ---------------------------------------------------------------------------
// Upsample einsum: up[t,b,e] = sum_s w[s,b,t] * enc[s,b,e] via f32 WMMA,
// 4 N-tiles per wave (A fragment reused 4x).
// grid=(ceil(T/64), 4, 64), block=128 (4 waves, wave per t-tile)
// ---------------------------------------------------------------------------
__global__ __launch_bounds__(128) void su_upsample_gemm(const float* __restrict__ wbuf,
                                                        const float* __restrict__ enc,
                                                        float* __restrict__ out, int T) {
    int b = blockIdx.z;
    int col0 = blockIdx.y * 64;          // 4 tiles of 16
    int lane = threadIdx.x & 31, wid = threadIdx.x >> 5;
    int row0 = (blockIdx.x * 4 + wid) * 16;
    if (row0 >= T) return;
    int lm = lane & 15;
    int kh = (lane >> 4) * 2;
    int mb = (lane >> 4) * 8;
    int tA = row0 + lm;
    bool okA = tA < T;
    v8f acc[4] = {};
    for (int k0 = 0; k0 < S_IN; k0 += 4) {
        int s0 = k0 + kh;
        v2f a;
        a.x = okA ? wbuf[(size_t)(s0 * BATCH + b) * T + tA] : 0.0f;
        a.y = okA ? wbuf[(size_t)((s0 + 1) * BATCH + b) * T + tA] : 0.0f;
        const float* er = enc + (size_t)(s0 * BATCH + b) * EMB + col0 + lm;
#pragma unroll
        for (int j = 0; j < 4; ++j) {
            v2f bb;
            bb.x = er[j * 16];
            bb.y = er[j * 16 + (size_t)BATCH * EMB];
            acc[j] = __builtin_amdgcn_wmma_f32_16x16x4_f32(false, a, false, bb, (short)0,
                                                           acc[j], false, false);
        }
    }
#pragma unroll
    for (int j = 0; j < 4; ++j)
        for (int v = 0; v < 8; ++v) {
            int tt = row0 + mb + v;
            if (tt < T) out[((size_t)tt * BATCH + b) * OUTE + col0 + j * 16 + lm] = acc[j][v];
        }
}

// ---------------------------------------------------------------------------
// Positional embedding gather (binary search into int cumsum)
// ---------------------------------------------------------------------------
__global__ void su_pos(const int* __restrict__ csumT, const int* __restrict__ total,
                       const float* __restrict__ encpos, float* __restrict__ out, int T) {
    int i = blockIdx.x * blockDim.x + threadIdx.x;
    if (i >= T * BATCH) return;
    int t = i >> 6, b = i & 63;
    const int* cs = csumT + b * S_IN;
    int lo = 0, hi = S_IN;
    while (lo < hi) { int mid = (lo + hi) >> 1; if (cs[mid] <= t) lo = mid + 1; else hi = mid; }
    int start = lo > 0 ? cs[lo - 1] : 0;
    int p = iclamp(t - start, 0, 999);
    bool ok = t < total[b];
    float* dst = out + ((size_t)t * BATCH + b) * OUTE + EMB;
    const float* src = encpos + p * POSD;
    for (int k = 0; k < POSD; ++k) dst[k] = ok ? src[k] : 0.0f;
}

// ---------------------------------------------------------------------------
// Host orchestration
// ---------------------------------------------------------------------------
extern "C" void kernel_launch(void* const* d_in, const int* in_sizes, int n_in,
                              void* d_out, int out_size, void* d_ws, size_t ws_size,
                              hipStream_t stream) {
    const float* enc       = (const float*)d_in[0];
    const float* durations = (const float*)d_in[1];
    const float* tempo     = (const float*)d_in[2];
    const int*   td        = (const int*)d_in[3];
    const int*   lens      = (const int*)d_in[4];
    const float* dWihf = (const float*)d_in[6];
    const float* dWhhf = (const float*)d_in[7];
    const float* dWihb = (const float*)d_in[8];
    const float* dWhhb = (const float*)d_in[9];
    const float* dproj = (const float*)d_in[10];
    const float* rWihf = (const float*)d_in[11];
    const float* rWhhf = (const float*)d_in[12];
    const float* rbihf = (const float*)d_in[13];
    const float* rbhhf = (const float*)d_in[14];
    const float* rWihb = (const float*)d_in[15];
    const float* rWhhb = (const float*)d_in[16];
    const float* rbihb = (const float*)d_in[17];
    const float* rbhhb = (const float*)d_in[18];
    const float* rproj = (const float*)d_in[19];
    const float* rpb   = (const float*)d_in[20];
    const float* posenc= (const float*)d_in[21];

    const int S = S_IN, B = BATCH;
    const int M = S * B;  // 24576
    // out_size = S*B + T*B*288 + B + S*B*T  ->  solve for T
    int T = (int)(((long long)out_size - (long long)M - B) /
                  ((long long)B * OUTE + (long long)M));

    // workspace carve-up (4-byte granules)
    float* w = (float*)d_ws;
    size_t off = 0;
    auto alloc = [&](size_t n) { float* p = w + off; off += n; return p; };
    float* Xd = alloc((size_t)M * KPAD);
    float* Xr = alloc((size_t)M * KPAD);
    float* WT[4];   for (int i = 0; i < 4; ++i) WT[i]   = alloc((size_t)KPAD * G4H);
    __bf16* WhP[4]; for (int i = 0; i < 4; ++i) WhP[i]  = (__bf16*)alloc((size_t)HID * G4H / 2);
    float* xg[4];   for (int i = 0; i < 4; ++i) xg[i]   = alloc((size_t)M * G4H);
    float* ob[4];   for (int i = 0; i < 4; ++i) ob[i]   = alloc((size_t)M * HID);
    float* biasf = alloc(G4H);
    float* biasb = alloc(G4H);
    float* ranges   = alloc((size_t)M);
    float* centersT = alloc((size_t)M);
    float* invT     = alloc((size_t)M);
    float* lgT      = alloc((size_t)M);
    int*   csumT = (int*)alloc((size_t)M);
    int*   total = (int*)alloc(B);

    float* o_pd  = (float*)d_out;
    float* o_out = o_pd + M;
    float* o_sl  = o_out + (size_t)T * B * OUTE;
    float* o_w   = o_sl + B;

    // --- prep ---
    {
        int n = M * KPAD;
        su_build_X<<<(n + 255) / 256, 256, 0, stream>>>(durations, tempo, td, enc, Xd, Xr, M);
    }
    const float* wih[4] = {dWihf, dWihb, rWihf, rWihb};
    const float* whh[4] = {dWhhf, dWhhb, rWhhf, rWhhb};
    for (int i = 0; i < 4; ++i) {
        su_build_WT<<<(KPAD * G4H + 255) / 256, 256, 0, stream>>>(wih[i], WT[i]);
        su_build_WhhP<<<(HID * G4H + 255) / 256, 256, 0, stream>>>(whh[i], WhP[i]);
    }
    su_build_bias<<<2, 256, 0, stream>>>(rbihf, rbhhf, biasf);
    su_build_bias<<<2, 256, 0, stream>>>(rbihb, rbhhb, biasb);

    // --- input GEMMs (4 problems in grid.z) ---
    GemmSet gset;
    gset.X[0] = Xd; gset.X[1] = Xd; gset.X[2] = Xr; gset.X[3] = Xr;
    for (int i = 0; i < 4; ++i) { gset.W[i] = WT[i]; gset.Y[i] = xg[i]; }
    gset.bias[0] = nullptr; gset.bias[1] = nullptr;
    gset.bias[2] = biasf;   gset.bias[3] = biasb;
    su_input_gemm<<<dim3(M / 16 / 8, G4H / 16 / 4, 4), 256, 0, stream>>>(gset, M);

    // --- recurrence (4 directions in parallel blocks, 272KB LDS each) ---
    RecSet rset;
    for (int i = 0; i < 4; ++i) {
        rset.xg[i] = xg[i]; rset.whhP[i] = WhP[i]; rset.outb[i] = ob[i];
    }
    rset.rev[0] = 0; rset.rev[1] = 1; rset.rev[2] = 0; rset.rev[3] = 1;
    su_lstm_rec<<<4, 1024, REC_SMEM, stream>>>(rset, lens);

    // --- projections ---
    su_proj<<<(M + 127) / 128, 128, 0, stream>>>(ob[0], ob[1], ob[2], ob[3],
                                                 dproj, rproj, rpb, lens, o_pd, ranges);

    // --- scans + range prep ---
    su_scan<<<1, 64, 0, stream>>>(td, centersT, csumT, total, o_sl);
    su_rng_prep<<<(M + 255) / 256, 256, 0, stream>>>(ranges, invT, lgT);

    // --- softmax weights into output ---
    su_softmax<<<dim3((T + 255) / 256, B), 256, 0, stream>>>(centersT, invT, lgT, o_w, T);

    // --- upsample GEMM into out[...,0:256] ---
    su_upsample_gemm<<<dim3((T + 63) / 64, 4, B), 128, 0, stream>>>(o_w, enc, o_out, T);

    // --- positional embedding into out[...,256:288] ---
    su_pos<<<(T * B + 255) / 256, 256, 0, stream>>>(csumT, total, posenc, o_out, T);
}
